// OffsetPredNet_69630009803203
// MI455X (gfx1250) — matrix operand order repaired and checked by reading, not compiled, not run on hardware
//
#include <hip/hip_runtime.h>
#include <hip/hip_bf16.h>
#include <cstdint>

typedef unsigned short u16;
typedef unsigned int   u32;

typedef __attribute__((ext_vector_type(16))) __bf16 v16bf;
typedef __attribute__((ext_vector_type(8)))  float  v8f;

#define NV   778   // vertices
#define KSP  9     // spiral length
#define NB   64    // batch B
#define MROW (NB * NV)  // 49792 = 389 * 128

#define BM   128
#define BN   128
#define BK   32
#define ASTR 40    // LDS row stride (u16 units): 16B-aligned, bank-conflict padding
#define BSTR 40

__device__ __forceinline__ u16 f32_to_bf16(float f) {
  u32 u = __float_as_uint(f);
  u += 0x7FFFu + ((u >> 16) & 1u);   // round-to-nearest-even
  return (u16)(u >> 16);
}
__device__ __forceinline__ float bf16_to_f32(u16 h) {
  return __uint_as_float(((u32)h) << 16);
}

// gfx1250 async global->LDS copy (ASYNCcnt-tracked, per-lane addresses).
// ldsaddr = LDS byte address (low 32 bits of generic shared pointer), 16B aligned.
__device__ __forceinline__ void async_copy_b128(u32 ldsaddr, const void* gaddr) {
  asm volatile("global_load_async_to_lds_b128 %0, %1, off"
               :: "v"(ldsaddr), "v"(gaddr) : "memory");
}
__device__ __forceinline__ void wait_asynccnt0() {
  asm volatile("s_wait_asynccnt 0x0" ::: "memory");
}

// ---------------- tiny prelude kernels ----------------

__global__ void init_fingermap(int* fmap) {
  int i = blockIdx.x * blockDim.x + threadIdx.x;
  if (i < NV) fmap[i] = -1;
}

__global__ void scatter_finger(int* fmap, const int* __restrict__ f, int n, int fid) {
  int i = blockIdx.x * blockDim.x + threadIdx.x;
  if (i < n) atomicMax(&fmap[f[i]], fid);   // higher finger id wins == sequential .set order
}

// pred_occ [192,2,5] -> occ[r,f] = softmax over axis=1, class 0 = sigmoid(p0-p1)
__global__ void occ_kernel(const float* __restrict__ p, float* __restrict__ occ) {
  int i = blockIdx.x * blockDim.x + threadIdx.x;
  if (i >= 192 * 5) return;
  int r = i / 5, f = i - r * 5;
  float p0 = p[r * 10 + f];
  float p1 = p[r * 10 + 5 + f];
  occ[i] = 1.0f / (1.0f + __expf(p1 - p0));
}

// h0[b,v,:] = [ sum_s softmax_s(w_s) * x[s*64+b,v,:] | x[b,v,:] ]  as bf16 (width 512)
__global__ __launch_bounds__(256) void prep_h0(const float* __restrict__ x,
                                               const float* __restrict__ occ,
                                               const int* __restrict__ fmap,
                                               u16* __restrict__ h0) {
  int bv = blockIdx.x;          // 0..MROW-1
  int c  = threadIdx.x;         // 0..255
  int b  = bv / NV;
  int v  = bv - b * NV;
  int f  = fmap[v];
  float w0 = 1.f, w1 = 1.f, w2 = 1.f;
  if (f >= 0) {
    w0 = occ[(b)       * 5 + f];
    w1 = occ[(64 + b)  * 5 + f];
    w2 = occ[(128 + b) * 5 + f];
  }
  float mx = fmaxf(w0, fmaxf(w1, w2));
  float e0 = __expf(w0 - mx), e1 = __expf(w1 - mx), e2 = __expf(w2 - mx);
  float inv = 1.0f / (e0 + e1 + e2);
  float xv0 = x[((size_t)(b)       * NV + v) * 256 + c];
  float xv1 = x[((size_t)(64 + b)  * NV + v) * 256 + c];
  float xv2 = x[((size_t)(128 + b) * NV + v) * 256 + c];
  float wx = (e0 * inv) * xv0 + (e1 * inv) * xv1 + (e2 * inv) * xv2;
  u16* row = h0 + (size_t)bv * 512;
  row[c]       = f32_to_bf16(wx);
  row[256 + c] = f32_to_bf16(xv0);
}

// w [Kdim, Cout] f32  ->  wt [Cout, Kdim] bf16 (transposed for contiguous B-fragment loads)
__global__ void convert_w_t(const float* __restrict__ w, u16* __restrict__ wt,
                            int Kdim, int Cout) {
  int i = blockIdx.x * blockDim.x + threadIdx.x;
  if (i >= Kdim * Cout) return;
  int n = i / Kdim, k = i - n * Kdim;
  wt[i] = f32_to_bf16(w[(size_t)k * Cout + n]);
}

// ---------------- WMMA spiral-conv GEMM ----------------
// out[m, n] = relu( sum_{k} hin[b(m), idx[v(m), k/Cin], k%Cin] * w[k, n] + bias[n] )
// 256 threads = 8 waves (4M x 2N); block tile 128x128; wave tile 32x64
// (2 m-subs x 4 n-subs = 8 x wmma 16x16x32 bf16 per K-step).
// Software-pipelined: async global->LDS copies for tile i+1 (ASYNCcnt) overlap
// WMMAs of tile i; LDS ping-pong, s_wait_asynccnt + one barrier per K-step.

union ABFrag {
  v16bf v;
  uint4 q[2];
};

__device__ __forceinline__ v8f wmma_bf16(const ABFrag& a, const ABFrag& b, v8f c) {
  return __builtin_amdgcn_wmma_f32_16x16x32_bf16(false, a.v, false, b.v,
                                                 (short)0, c, false, false);
}

__global__ __launch_bounds__(256) void spiral_gemm_bf16(
    const u16* __restrict__ hin,      // [MROW, Cin] bf16
    const u16* __restrict__ wt,       // [Cout, Kdim] bf16 (transposed)
    const int* __restrict__ idxp,     // [NV, KSP]
    const float* __restrict__ bias,   // [Cout]
    u16* __restrict__ hout,           // [MROW, Cout] bf16
    int lcin, int Kdim, int Cout, int relu)
{
  __shared__ u16 aT[2][BM * ASTR];   // 2 x 10240 B
  __shared__ u16 bT[2][BN * BSTR];   // 2 x 10240 B
  __shared__ int nbr[BM * KSP];
  __shared__ int bBase[BM];

  const int tid   = threadIdx.x;
  const int mbase = blockIdx.y * BM;
  const int nbase = blockIdx.x * BN;
  const int cmask = (1 << lcin) - 1;

  // per-row batch base + spiral neighbor table for this block
  for (int t = tid; t < BM; t += 256) {
    int m = mbase + t;
    bBase[t] = (m / NV) * NV;
  }
  for (int t = tid; t < BM * KSP; t += 256) {
    int r = t / KSP, j = t - r * KSP;
    int v = (mbase + r) % NV;
    nbr[t] = idxp[v * KSP + j];
  }
  __syncthreads();

  // A loader: 128 rows x 32 k -> 2 x b128 per thread (gathered)
  const int aRow = tid >> 1;            // 0..127
  const int aKK0 = (tid & 1) << 3;      // 0 or 8
  const int aKK1 = aKK0 + 16;           // 16 or 24
  // B loader: 128 n x 32 k -> 2 x b128 per thread (dense, pre-transposed weights)
  const int bRow = tid >> 1;            // 0..127
  const int bKK0 = (tid & 1) << 3;
  const int bKK1 = bKK0 + 16;

  // per-thread LDS byte addresses for the async copies (buffer 0; buffer 1 at +stride)
  const u32 aLds0 = (u32)(size_t)&aT[0][aRow * ASTR + aKK0];
  const u32 aLds1 = (u32)(size_t)&aT[0][aRow * ASTR + aKK1];
  const u32 bLds0 = (u32)(size_t)&bT[0][bRow * BSTR + bKK0];
  const u32 bLds1 = (u32)(size_t)&bT[0][bRow * BSTR + bKK1];
  const u32 aBufStride = (u32)(BM * ASTR * 2);
  const u32 bBufStride = (u32)(BN * BSTR * 2);

  auto asyncLoadTile = [&](int buf, int kt) {
    u32 abuf = (u32)buf * aBufStride;
    u32 bbuf = (u32)buf * bBufStride;
    {
      int kidx = kt + aKK0;
      int j = kidx >> lcin, c = kidx & cmask;
      int nb = nbr[aRow * KSP + j];
      async_copy_b128(aLds0 + abuf, hin + (((size_t)(bBase[aRow] + nb)) << lcin) + c);
    }
    {
      int kidx = kt + aKK1;
      int j = kidx >> lcin, c = kidx & cmask;
      int nb = nbr[aRow * KSP + j];
      async_copy_b128(aLds1 + abuf, hin + (((size_t)(bBase[aRow] + nb)) << lcin) + c);
    }
    const u16* wrow = wt + (size_t)(nbase + bRow) * Kdim + kt;
    async_copy_b128(bLds0 + bbuf, wrow + bKK0);
    async_copy_b128(bLds1 + bbuf, wrow + bKK1);
  };

  const int wid   = tid >> 5;
  const int lane  = tid & 31;
  const int waveM = wid & 3;            // 4 M-slices of 32 rows
  const int waveN = wid >> 2;           // 2 N-slices of 64 cols
  const int l15   = lane & 15;
  const int kh    = lane >> 4;

  v8f acc[2][4];
#pragma unroll
  for (int mi = 0; mi < 2; ++mi)
#pragma unroll
    for (int ns = 0; ns < 4; ++ns) acc[mi][ns] = (v8f){};

  // prologue: fill buffer 0
  asyncLoadTile(0, 0);
  wait_asynccnt0();
  __syncthreads();

  const int nk = Kdim / BK;             // 144 or 72
  const int arBase = (waveM * 32 + l15) * ASTR;        // m-sub 0; +16*ASTR for m-sub 1
  const int brBase = (waveN * 64 + l15) * BSTR;        // n-sub 0; +16*BSTR per n-sub

  for (int it = 0; it < nk; ++it) {
    const int cur = it & 1;
    if (it + 1 < nk) asyncLoadTile(1 - cur, (it + 1) * BK);  // DMA next tile into other buffer

    // A fragments (ISA 16-bit A 16x32 layout): lanes 0-15 K=0..7 & 16..23,
    // lanes 16-31 K=8..15 & 24..31
    ABFrag a0, a1, bf[4];
    a0.q[0] = *(const uint4*)&aT[cur][arBase + kh * 8];
    a0.q[1] = *(const uint4*)&aT[cur][arBase + 16 + kh * 8];
    a1.q[0] = *(const uint4*)&aT[cur][arBase + 16 * ASTR + kh * 8];
    a1.q[1] = *(const uint4*)&aT[cur][arBase + 16 * ASTR + 16 + kh * 8];
    // B fragments: N = lane&15, same K layout as A (lanes 0-15 K=0..7 & 16..23, lanes 16-31 K=8..15 & 24..31)
#pragma unroll
    for (int ns = 0; ns < 4; ++ns) {
      int br = brBase + ns * 16 * BSTR;
      bf[ns].q[0] = *(const uint4*)&bT[cur][br + kh * 8];
      bf[ns].q[1] = *(const uint4*)&bT[cur][br + 16 + kh * 8];
    }

#pragma unroll
    for (int ns = 0; ns < 4; ++ns) {
      acc[0][ns] = wmma_bf16(a0, bf[ns], acc[0][ns]);
      acc[1][ns] = wmma_bf16(a1, bf[ns], acc[1][ns]);
    }

    if (it + 1 < nk) wait_asynccnt0();            // next tile landed in LDS
    __syncthreads();
  }

  // epilogue: C/D layout -> M = (lane>>4)*8 + i, N = lane&15
#pragma unroll
  for (int ns = 0; ns < 4; ++ns) {
    int ng = nbase + waveN * 64 + ns * 16 + l15;
    float bb = bias[ng];
#pragma unroll
    for (int mi = 0; mi < 2; ++mi) {
#pragma unroll
      for (int i = 0; i < 8; ++i) {
        int mg = mbase + waveM * 32 + mi * 16 + kh * 8 + i;
        float vv = acc[mi][ns][i] + bb;
        if (relu) vv = fmaxf(vv, 0.f);
        hout[(size_t)mg * Cout + ng] = f32_to_bf16(vv);
      }
    }
  }
}

// ---------------- head: N=3, plain fp32 reduction ----------------
__global__ __launch_bounds__(128) void head_kernel(const u16* __restrict__ hin,  // [MROW,256] bf16
                                                   const int* __restrict__ idxp,
                                                   const float* __restrict__ wh, // [2304,3]
                                                   const float* __restrict__ bh, // [3]
                                                   float* __restrict__ out)      // [MROW,3]
{
  int m = blockIdx.x;
  int tid = threadIdx.x;
  int b = m / NV;
  int v = m - b * NV;
  size_t base = (size_t)b * NV;
  float a0 = 0.f, a1 = 0.f, a2 = 0.f;
  for (int k = tid; k < 2304; k += 128) {
    int j = k >> 8;
    int c = k & 255;
    int nb = idxp[v * KSP + j];
    float a = bf16_to_f32(hin[((base + nb) << 8) + c]);
    a0 += a * wh[k * 3 + 0];
    a1 += a * wh[k * 3 + 1];
    a2 += a * wh[k * 3 + 2];
  }
  __shared__ float s0[128], s1[128], s2[128];
  s0[tid] = a0; s1[tid] = a1; s2[tid] = a2;
  __syncthreads();
  for (int off = 64; off > 0; off >>= 1) {
    if (tid < off) { s0[tid] += s0[tid + off]; s1[tid] += s1[tid + off]; s2[tid] += s2[tid + off]; }
    __syncthreads();
  }
  if (tid == 0) {
    out[(size_t)m * 3 + 0] = s0[0] + bh[0];
    out[(size_t)m * 3 + 1] = s1[0] + bh[1];
    out[(size_t)m * 3 + 2] = s2[0] + bh[2];
  }
}

// ---------------- launch ----------------
extern "C" void kernel_launch(void* const* d_in, const int* in_sizes, int n_in,
                              void* d_out, int out_size, void* d_ws, size_t ws_size,
                              hipStream_t stream) {
  const float* x        = (const float*)d_in[0];
  const float* pred_occ = (const float*)d_in[1];
  const int*   indices  = (const int*)d_in[2];
  const int*   fingers[5] = { (const int*)d_in[3], (const int*)d_in[4], (const int*)d_in[5],
                              (const int*)d_in[6], (const int*)d_in[7] };
  const float* w0 = (const float*)d_in[8];
  const float* b0 = (const float*)d_in[9];
  const float* w1 = (const float*)d_in[10];
  const float* b1 = (const float*)d_in[11];
  const float* w2 = (const float*)d_in[12];
  const float* b2 = (const float*)d_in[13];
  const float* wh = (const float*)d_in[14];
  const float* bh = (const float*)d_in[15];

  char* ws = (char*)d_ws;
  constexpr size_t M = (size_t)MROW;
  size_t offA   = 0;                              // activation ping buffer (width <= 512 bf16)
  size_t offB   = offA  + M * 512 * 2;            // activation pong buffer
  size_t offW0  = offB  + M * 512 * 2;
  size_t offW1  = offW0 + (size_t)4608 * 512 * 2;
  size_t offW2  = offW1 + (size_t)4608 * 256 * 2;
  size_t offOcc = offW2 + (size_t)2304 * 256 * 2;
  size_t offFm  = offOcc + 192 * 5 * 4;

  u16*   bufA = (u16*)(ws + offA);
  u16*   bufB = (u16*)(ws + offB);
  u16*   w0t  = (u16*)(ws + offW0);
  u16*   w1t  = (u16*)(ws + offW1);
  u16*   w2t  = (u16*)(ws + offW2);
  float* occ  = (float*)(ws + offOcc);
  int*   fmap = (int*)(ws + offFm);

  init_fingermap<<<(NV + 255) / 256, 256, 0, stream>>>(fmap);
  for (int i = 0; i < 5; ++i) {
    int n = in_sizes[3 + i];
    scatter_finger<<<(n + 255) / 256, 256, 0, stream>>>(fmap, fingers[i], n, i);
  }
  occ_kernel<<<(192 * 5 + 255) / 256, 256, 0, stream>>>(pred_occ, occ);
  prep_h0<<<MROW, 256, 0, stream>>>(x, occ, fmap, bufA);

  convert_w_t<<<((4608 * 512) + 255) / 256, 256, 0, stream>>>(w0, w0t, 4608, 512);
  convert_w_t<<<((4608 * 256) + 255) / 256, 256, 0, stream>>>(w1, w1t, 4608, 256);
  convert_w_t<<<((2304 * 256) + 255) / 256, 256, 0, stream>>>(w2, w2t, 2304, 256);

  // L0: [M,4608] x [4608,512] -> relu -> bufB
  spiral_gemm_bf16<<<dim3(512 / BN, MROW / BM), 256, 0, stream>>>(
      bufA, w0t, indices, b0, bufB, 9, 4608, 512, 1);
  // L1: [M,4608] x [4608,256] -> relu -> bufA
  spiral_gemm_bf16<<<dim3(256 / BN, MROW / BM), 256, 0, stream>>>(
      bufB, w1t, indices, b1, bufA, 9, 4608, 256, 1);
  // L2: [M,2304] x [2304,256] -> relu -> bufB
  spiral_gemm_bf16<<<dim3(256 / BN, MROW / BM), 256, 0, stream>>>(
      bufA, w2t, indices, b2, bufB, 8, 2304, 256, 1);
  // head: [M,2304] x [2304,3] fp32 -> d_out
  head_kernel<<<MROW, 128, 0, stream>>>(bufB, indices, wh, bh, (float*)d_out);
}
